// XFormersMHA_IP_59554016526851
// MI455X (gfx1250) — compile-verified, matrix operand order checked by reading, not run on hardware
//
#include <hip/hip_runtime.h>
#include <hip/hip_bf16.h>
#include <stdint.h>

// ---------------------------------------------------------------------------
// Types for CDNA5 WMMA (wave32): 16x16x32 bf16 -> f32
// ---------------------------------------------------------------------------
typedef __bf16 v16bf __attribute__((ext_vector_type(16)));
typedef float  v8f   __attribute__((ext_vector_type(8)));
typedef unsigned int u32x4 __attribute__((ext_vector_type(4)));

#define DEV static __device__ __forceinline__

DEV unsigned short f2bf(float f) {
  union { float f; unsigned int u; } a; a.f = f;
  unsigned int u = a.u;
  unsigned int r = u + 0x7FFFu + ((u >> 16) & 1u);  // round-to-nearest-even
  return (unsigned short)(r >> 16);
}

union Frag16 { v16bf v; u32x4 q[2]; unsigned short s[16]; };

// A/B fragment for 16x32 bf16 (ISA 7.12.2): lane l(0..15) row M=l holds
// K={0..7,16..23}; lane l+16 holds K={8..15,24..31}. p = row start (16B align).
DEV void ldfrag(Frag16& f, const unsigned short* p, int hi) {
  f.q[0] = *(const u32x4*)(p + hi * 8);
  f.q[1] = *(const u32x4*)(p + 16 + hi * 8);
}

DEV v8f wmma_bf16(const Frag16& a, const Frag16& b, v8f c) {
  return __builtin_amdgcn_wmma_f32_16x16x32_bf16(false, a.v, false, b.v,
                                                 (short)0, c, false, false);
}

// CDNA5 async memory->LDS (GV mode): per-lane 16B, tracked by ASYNCcnt.
// lds_off = wave-relative LDS byte offset (flat shared addr low 32 bits).
DEV void async_ld16(unsigned lds_off, const unsigned short* gptr) {
  unsigned long long ga = (unsigned long long)gptr;
  asm volatile("global_load_async_to_lds_b128 %0, %1, off"
               :: "v"(lds_off), "v"(ga) : "memory");
}
DEV void wait_async0() { asm volatile("s_wait_asynccnt 0x0" ::: "memory"); }

// ---------------------------------------------------------------------------
// fp32 -> bf16 elementwise (weights + activations)
// ---------------------------------------------------------------------------
__global__ __launch_bounds__(256) void cvt_bf16_kernel(
    const float* __restrict__ in, unsigned short* __restrict__ out, int n) {
  int i = blockIdx.x * 256 + threadIdx.x;
  if (i < n) out[i] = f2bf(in[i]);
}

// ---------------------------------------------------------------------------
// Tiled WMMA GEMM: C[M,N] = A[M,K] * B[N,K]^T (+bias).  A,B bf16.
// Block 256 thr = 8 waves; tile 128x128, K-step 32; double-buffered LDS
// staged entirely with global_load_async_to_lds_b128 (overlaps WMMA).
// aBatch/cBatch in ELEMENTS; grid.z = batch. N multiple of 128; K mult of 32.
// ---------------------------------------------------------------------------
template <bool OUT_F32_BIAS>
__global__ __launch_bounds__(256) void gemm_wmma_kernel(
    const unsigned short* __restrict__ Am, const unsigned short* __restrict__ Bm,
    const float* __restrict__ bias, void* __restrict__ Cop,
    int M, int K, int lda, int ldb, int ldc, long aBatch, long cBatch) {
  // [buf][ A 128x40 | B 128x40 ] bf16; stride 40 (+8 pad) -> conflict-free.
  __shared__ unsigned short S[2 * 10240];
  const unsigned ldsBase = (unsigned)(size_t)&S[0];

  const int tid = threadIdx.x;
  const int wid = tid >> 5, lane = tid & 31;
  const int l = lane & 15, hi = lane >> 4;
  const int wm = wid & 3, wn = wid >> 2;  // 4x2 wave grid: 32x64 per wave
  const int m0 = blockIdx.y * 128;
  const int n0 = blockIdx.x * 128;
  const long zA = (long)blockIdx.z * aBatch;
  const long zC = (long)blockIdx.z * cBatch;

  v8f acc[2][4];
#pragma unroll
  for (int i = 0; i < 2; i++)
#pragma unroll
    for (int j = 0; j < 4; j++)
#pragma unroll
      for (int e = 0; e < 8; e++) acc[i][j][e] = 0.0f;

  const int lr = tid >> 1;        // 0..127 tile row
  const int lc = (tid & 1) * 16;  // 0/16: each thread stages 32B of A and B

  int arow = m0 + lr;
  if (arow > M - 1) arow = M - 1;  // clamp; rows >= M never stored
  const unsigned short* agBase = Am + zA + (long)arow * lda + lc;
  const unsigned short* bgBase = Bm + (long)(n0 + lr) * ldb + lc;
  const unsigned aoffBase = ldsBase + (unsigned)(lr * 40 + lc) * 2;

  const int KT = K >> 5;

  auto stage = [&](int kt, int buf) {
    const int k0 = kt << 5;
    const unsigned short* ag = agBase + k0;
    const unsigned short* bg = bgBase + k0;
    const unsigned aoff = aoffBase + (unsigned)buf * 20480u;
    const unsigned boff = aoff + 10240u;
    async_ld16(aoff, ag);
    async_ld16(aoff + 16, ag + 8);
    async_ld16(boff, bg);
    async_ld16(boff + 16, bg + 8);
  };

  stage(0, 0);
  wait_async0();
  __syncthreads();

  for (int kt = 0; kt < KT; ++kt) {
    const int buf = kt & 1;
    if (kt + 1 < KT) stage(kt + 1, buf ^ 1);  // overlap with WMMA below

    const unsigned short* Ab = &S[buf * 10240];
    const unsigned short* Bb = Ab + 5120;
    Frag16 am[2], bn[4];
#pragma unroll
    for (int i = 0; i < 2; i++) ldfrag(am[i], &Ab[(wm * 32 + i * 16 + l) * 40], hi);
#pragma unroll
    for (int j = 0; j < 4; j++) ldfrag(bn[j], &Bb[(wn * 64 + j * 16 + l) * 40], hi);
#pragma unroll
    for (int i = 0; i < 2; i++)
#pragma unroll
      for (int j = 0; j < 4; j++) acc[i][j] = wmma_bf16(am[i], bn[j], acc[i][j]);

    wait_async0();    // next-buffer DMA complete (own wave)
    __syncthreads();  // all waves done reading cur buf / writing next buf
  }

  // C layout: VGPR r -> M = r + 8*hi ; N = lane&15
#pragma unroll
  for (int i = 0; i < 2; i++) {
#pragma unroll
    for (int j = 0; j < 4; j++) {
      const int rowb = m0 + wm * 32 + i * 16 + hi * 8;
      const int col = n0 + wn * 64 + j * 16 + l;
#pragma unroll
      for (int r = 0; r < 8; r++) {
        const int row = rowb + r;
        if (row < M) {
          if (OUT_F32_BIAS) {
            ((float*)Cop)[zC + (long)row * ldc + col] = acc[i][j][r] + bias[col];
          } else {
            ((unsigned short*)Cop)[zC + (long)row * ldc + col] = f2bf(acc[i][j][r]);
          }
        }
      }
    }
  }
}

// ---------------------------------------------------------------------------
// V transpose: src [B][rowsAlloc][1280] bf16 -> dst [B][20][64][kpad] bf16
// (zero pad keys >= rowsValid so P·V WMMA over padded K is exact)
// ---------------------------------------------------------------------------
__global__ __launch_bounds__(256) void transpose_v_kernel(
    const unsigned short* __restrict__ src, unsigned short* __restrict__ dst,
    int rowsValid, int kpad, int rowsAlloc) {
  int idx = blockIdx.x * 256 + threadIdx.x;
  int kp = idx % kpad;
  int t = idx / kpad;
  int d = t % 64; t /= 64;
  int h = t % 20;
  int b = t / 20;
  unsigned short v = 0;
  if (kp < rowsValid) v = src[((long)b * rowsAlloc + kp) * 1280 + h * 64 + d];
  dst[idx] = v;
}

// ---------------------------------------------------------------------------
// Fused dual attention: one wave per (b, h, 16-query tile).
// ---------------------------------------------------------------------------
__global__ __launch_bounds__(128) void attn_fused_kernel(
    const unsigned short* __restrict__ Q,     // [b][4096][1280]
    const unsigned short* __restrict__ Ktxt,  // [b][80][1280] (rows>=77 junk)
    const unsigned short* __restrict__ Kimg,  // [b][16][1280]
    const unsigned short* __restrict__ Vtt,   // [b][h][64][96] zero-padded
    const unsigned short* __restrict__ Vti,   // [b][h][64][32] zero-padded
    const float* __restrict__ scale_ptr,
    unsigned short* __restrict__ Out)         // [b][4096][1280]
{
  __shared__ unsigned short Pt[4][16 * 104];  // txt probs, stride 104 (pad)
  __shared__ unsigned short Pi[4][16 * 40];   // img probs, stride 40 (pad)
  const int wid = threadIdx.x >> 5, lane = threadIdx.x & 31;
  const int l = lane & 15, hi = lane >> 4;
  const int b = blockIdx.z, h = blockIdx.y;
  const int q0 = (blockIdx.x * 4 + wid) * 16;

  const unsigned short* Qb = Q + ((long)b * 4096 + q0) * 1280 + h * 64;
  Frag16 qa[2];
#pragma unroll
  for (int kf = 0; kf < 2; kf++) ldfrag(qa[kf], Qb + (long)l * 1280 + kf * 32, hi);

  v8f At[5], Ai;
#pragma unroll
  for (int nt = 0; nt < 5; nt++)
#pragma unroll
    for (int e = 0; e < 8; e++) At[nt][e] = 0.0f;
#pragma unroll
  for (int e = 0; e < 8; e++) Ai[e] = 0.0f;

  const unsigned short* Kb = Ktxt + (long)b * 80 * 1280 + h * 64;
#pragma unroll
  for (int nt = 0; nt < 5; nt++) {
    Frag16 k0f, k1f;
    ldfrag(k0f, Kb + (long)(nt * 16 + l) * 1280, hi);
    ldfrag(k1f, Kb + (long)(nt * 16 + l) * 1280 + 32, hi);
    At[nt] = wmma_bf16(qa[0], k0f, At[nt]);
    At[nt] = wmma_bf16(qa[1], k1f, At[nt]);
  }
  {
    const unsigned short* Kib = Kimg + (long)b * 16 * 1280 + h * 64;
    Frag16 k0f, k1f;
    ldfrag(k0f, Kib + (long)l * 1280, hi);
    ldfrag(k1f, Kib + (long)l * 1280 + 32, hi);
    Ai = wmma_bf16(qa[0], k0f, Ai);
    Ai = wmma_bf16(qa[1], k1f, Ai);
  }

  const float sc = 0.125f;  // 1/sqrt(64)

  // ---- txt softmax over 77 keys (cols 77..79 masked) ----
  float rm[8], rs[8];
#pragma unroll
  for (int e = 0; e < 8; e++) rm[e] = -3.0e38f;
#pragma unroll
  for (int nt = 0; nt < 5; nt++) {
    const bool valid = (nt * 16 + l) < 77;
#pragma unroll
    for (int e = 0; e < 8; e++) {
      float v = valid ? At[nt][e] * sc : -3.0e38f;
      At[nt][e] = v;
      rm[e] = fmaxf(rm[e], v);
    }
  }
#pragma unroll
  for (int e = 0; e < 8; e++) {
    rm[e] = fmaxf(rm[e], __shfl_xor(rm[e], 1));
    rm[e] = fmaxf(rm[e], __shfl_xor(rm[e], 2));
    rm[e] = fmaxf(rm[e], __shfl_xor(rm[e], 4));
    rm[e] = fmaxf(rm[e], __shfl_xor(rm[e], 8));
    rs[e] = 0.0f;
  }
#pragma unroll
  for (int nt = 0; nt < 5; nt++)
#pragma unroll
    for (int e = 0; e < 8; e++) {
      float p = __expf(At[nt][e] - rm[e]);
      At[nt][e] = p;
      rs[e] += p;
    }
#pragma unroll
  for (int e = 0; e < 8; e++) {
    rs[e] += __shfl_xor(rs[e], 1);
    rs[e] += __shfl_xor(rs[e], 2);
    rs[e] += __shfl_xor(rs[e], 4);
    rs[e] += __shfl_xor(rs[e], 8);
    rs[e] = 1.0f / rs[e];
  }
#pragma unroll
  for (int nt = 0; nt < 5; nt++)
#pragma unroll
    for (int e = 0; e < 8; e++)
      Pt[wid][(e + 8 * hi) * 104 + nt * 16 + l] = f2bf(At[nt][e] * rs[e]);
#pragma unroll
  for (int e = 0; e < 8; e++) Pt[wid][(e + 8 * hi) * 104 + 80 + l] = 0;

  // ---- img softmax over exactly 16 keys ----
  float im[8], is[8];
#pragma unroll
  for (int e = 0; e < 8; e++) {
    float v = Ai[e] * sc;
    Ai[e] = v;
    im[e] = v;
  }
#pragma unroll
  for (int e = 0; e < 8; e++) {
    im[e] = fmaxf(im[e], __shfl_xor(im[e], 1));
    im[e] = fmaxf(im[e], __shfl_xor(im[e], 2));
    im[e] = fmaxf(im[e], __shfl_xor(im[e], 4));
    im[e] = fmaxf(im[e], __shfl_xor(im[e], 8));
  }
#pragma unroll
  for (int e = 0; e < 8; e++) {
    float p = __expf(Ai[e] - im[e]);
    Ai[e] = p;
    is[e] = p;
  }
#pragma unroll
  for (int e = 0; e < 8; e++) {
    is[e] += __shfl_xor(is[e], 1);
    is[e] += __shfl_xor(is[e], 2);
    is[e] += __shfl_xor(is[e], 4);
    is[e] += __shfl_xor(is[e], 8);
    is[e] = 1.0f / is[e];
  }
#pragma unroll
  for (int e = 0; e < 8; e++) {
    Pi[wid][(e + 8 * hi) * 40 + l] = f2bf(Ai[e] * is[e]);
    Pi[wid][(e + 8 * hi) * 40 + 16 + l] = 0;  // zero-pad K 16..31
  }

  asm volatile("s_wait_dscnt 0x0" ::: "memory");  // same-wave LDS W->R order

  // ---- P·V ----
  v8f Ot[4], Oi4[4];
#pragma unroll
  for (int nt = 0; nt < 4; nt++)
#pragma unroll
    for (int e = 0; e < 8; e++) { Ot[nt][e] = 0.0f; Oi4[nt][e] = 0.0f; }

  const unsigned short* Vb = Vtt + ((long)(b * 20 + h)) * 64 * 96;
#pragma unroll
  for (int kf = 0; kf < 3; kf++) {
    Frag16 pa;
    ldfrag(pa, &Pt[wid][l * 104 + kf * 32], hi);
#pragma unroll
    for (int nt = 0; nt < 4; nt++) {
      Frag16 vb;
      ldfrag(vb, Vb + (long)(nt * 16 + l) * 96 + kf * 32, hi);
      Ot[nt] = wmma_bf16(pa, vb, Ot[nt]);
    }
  }
  {
    const unsigned short* Vib = Vti + ((long)(b * 20 + h)) * 64 * 32;
    Frag16 pa;
    ldfrag(pa, &Pi[wid][l * 40], hi);
#pragma unroll
    for (int nt = 0; nt < 4; nt++) {
      Frag16 vb;
      ldfrag(vb, Vib + (long)(nt * 16 + l) * 32, hi);
      Oi4[nt] = wmma_bf16(pa, vb, Oi4[nt]);
    }
  }

  const float s = scale_ptr[0];
  unsigned short* Ob = Out + ((long)b * 4096 + q0) * 1280 + h * 64;
#pragma unroll
  for (int nt = 0; nt < 4; nt++)
#pragma unroll
    for (int e = 0; e < 8; e++)
      Ob[(long)(e + 8 * hi) * 1280 + nt * 16 + l] = f2bf(Ot[nt][e] + s * Oi4[nt][e]);
}

// ---------------------------------------------------------------------------
// Host launch. Inputs: 0:x 1:context 2:scale 3:num_img_token(=16 hardcoded;
// eos=77) 4:Wq 5:Wk 6:Wv 7:Wk_ip 8:Wv_ip 9:Wout 10:b_out. WS ~285 MB.
// ---------------------------------------------------------------------------
extern "C" void kernel_launch(void* const* d_in, const int* in_sizes, int n_in,
                              void* d_out, int out_size, void* d_ws,
                              size_t ws_size, hipStream_t stream) {
  const float* x = (const float*)d_in[0];
  const float* ctx = (const float*)d_in[1];
  const float* scale = (const float*)d_in[2];
  const float* Wq = (const float*)d_in[4];
  const float* Wk = (const float*)d_in[5];
  const float* Wv = (const float*)d_in[6];
  const float* Wki = (const float*)d_in[7];
  const float* Wvi = (const float*)d_in[8];
  const float* Wo = (const float*)d_in[9];
  const float* bo = (const float*)d_in[10];
  float* out = (float*)d_out;

  char* ws = (char*)d_ws;
  size_t off = 0;
  auto take = [&](size_t bytes) -> unsigned short* {
    char* p = ws + off;
    off += (bytes + 255) & ~(size_t)255;
    return (unsigned short*)p;
  };
  unsigned short* WqB = take((size_t)1280 * 1280 * 2);
  unsigned short* WkB = take((size_t)1280 * 2048 * 2);
  unsigned short* WvB = take((size_t)1280 * 2048 * 2);
  unsigned short* WkiB = take((size_t)1280 * 2048 * 2);
  unsigned short* WviB = take((size_t)1280 * 2048 * 2);
  unsigned short* WoB = take((size_t)1280 * 1280 * 2);
  unsigned short* xB = take((size_t)8 * 4096 * 1280 * 2);
  unsigned short* ctxB = take((size_t)8 * 93 * 2048 * 2);
  unsigned short* Qb = take((size_t)8 * 4096 * 1280 * 2);
  unsigned short* Ktx = take((size_t)8 * 80 * 1280 * 2);
  unsigned short* Vtx = take((size_t)8 * 80 * 1280 * 2);
  unsigned short* Kim = take((size_t)8 * 16 * 1280 * 2);
  unsigned short* Vim = take((size_t)8 * 16 * 1280 * 2);
  unsigned short* Vtt = take((size_t)8 * 20 * 64 * 96 * 2);
  unsigned short* Vti = take((size_t)8 * 20 * 64 * 32 * 2);
  unsigned short* Att = take((size_t)8 * 4096 * 1280 * 2);
  if (off > ws_size) return;  // workspace too small; nothing safe to do

  // fp32 -> bf16 downcasts (weights + activations; enables pure-async GEMM)
  cvt_bf16_kernel<<<6400, 256, 0, stream>>>(Wq, WqB, 1280 * 1280);
  cvt_bf16_kernel<<<10240, 256, 0, stream>>>(Wk, WkB, 1280 * 2048);
  cvt_bf16_kernel<<<10240, 256, 0, stream>>>(Wv, WvB, 1280 * 2048);
  cvt_bf16_kernel<<<10240, 256, 0, stream>>>(Wki, WkiB, 1280 * 2048);
  cvt_bf16_kernel<<<10240, 256, 0, stream>>>(Wvi, WviB, 1280 * 2048);
  cvt_bf16_kernel<<<6400, 256, 0, stream>>>(Wo, WoB, 1280 * 1280);
  cvt_bf16_kernel<<<163840, 256, 0, stream>>>(x, xB, 8 * 4096 * 1280);
  cvt_bf16_kernel<<<5952, 256, 0, stream>>>(ctx, ctxB, 8 * 93 * 2048);

  // Q = x @ Wq^T   (M=32768, K=1280)
  gemm_wmma_kernel<false><<<dim3(10, 256, 1), 256, 0, stream>>>(
      xB, WqB, nullptr, Qb, 32768, 1280, 1280, 1280, 1280, 0L, 0L);
  // K/V txt: per-batch M=77 over context[:, :77, :]  (K=2048)
  gemm_wmma_kernel<false><<<dim3(10, 1, 8), 256, 0, stream>>>(
      ctxB, WkB, nullptr, Ktx, 77, 2048, 2048, 2048, 1280, (long)93 * 2048,
      (long)80 * 1280);
  gemm_wmma_kernel<false><<<dim3(10, 1, 8), 256, 0, stream>>>(
      ctxB, WvB, nullptr, Vtx, 77, 2048, 2048, 2048, 1280, (long)93 * 2048,
      (long)80 * 1280);
  // K/V img: per-batch M=16 over context[:, 77:, :]
  gemm_wmma_kernel<false><<<dim3(10, 1, 8), 256, 0, stream>>>(
      ctxB + (long)77 * 2048, WkiB, nullptr, Kim, 16, 2048, 2048, 2048, 1280,
      (long)93 * 2048, (long)16 * 1280);
  gemm_wmma_kernel<false><<<dim3(10, 1, 8), 256, 0, stream>>>(
      ctxB + (long)77 * 2048, WviB, nullptr, Vim, 16, 2048, 2048, 2048, 1280,
      (long)93 * 2048, (long)16 * 1280);

  // V -> [b][h][64][Kpad] (zero-padded) for P·V B-fragments
  transpose_v_kernel<<<3840, 256, 0, stream>>>(Vtx, Vtt, 77, 96, 80);
  transpose_v_kernel<<<1280, 256, 0, stream>>>(Vim, Vti, 16, 32, 16);

  // Fused dual attention
  attn_fused_kernel<<<dim3(64, 20, 8), 128, 0, stream>>>(Qb, Ktx, Kim, Vtt,
                                                         Vti, scale, Att);

  // out = attn @ Wout^T + b_out  (fp32 out)
  gemm_wmma_kernel<true><<<dim3(10, 256, 1), 256, 0, stream>>>(
      Att, WoB, bo, out, 32768, 1280, 1280, 1280, 1280, 0L, 0L);
}
